// Attention_38027640438938
// MI455X (gfx1250) — matrix operand order repaired and checked by reading, not compiled, not run on hardware
//
#include <hip/hip_runtime.h>
#include <stdint.h>

typedef __attribute__((ext_vector_type(16))) __bf16 v16bf;
typedef __attribute__((ext_vector_type(8)))  float  v8f;

#define S_LEN  1024
#define DMODEL 1024
#define NHEAD  16
#define DHEAD  64
#define BATCH  8
#define D3     3072

union FragU { v16bf v; uint4 q[2]; };

// Immediate-pattern XOR lane swap within 16-lane groups (group-of-32 ds_swizzle:
// new_lane = ((lane & 0x1f) | 0) ^ XOR). No address VGPR needed, unlike bpermute.
template <int XOR>
__device__ __forceinline__ float swz_xor(float x) {
    constexpr int pat = (XOR << 10) | 0x1f;
    return __int_as_float(__builtin_amdgcn_ds_swizzle(__float_as_int(x), pat));
}
__device__ __forceinline__ float red_max16(float x) {
    x = fmaxf(x, swz_xor<8>(x));
    x = fmaxf(x, swz_xor<4>(x));
    x = fmaxf(x, swz_xor<2>(x));
    x = fmaxf(x, swz_xor<1>(x));
    return x;
}
__device__ __forceinline__ float red_sum16(float x) {
    x += swz_xor<8>(x);
    x += swz_xor<4>(x);
    x += swz_xor<2>(x);
    x += swz_xor<1>(x);
    return x;
}

// A-matrix 16x32 bf16 fragment from LDS (row-major, stride ld elems).
// Layout: lane holds row M=lane%16; elements are K = 16*(lane/16) + e (contiguous, matches B).
__device__ __forceinline__ v16bf ld_frag_a(const __bf16* base, int ld, int lane) {
    const int row = lane & 15, half = lane >> 4;
    const __bf16* p = base + row * ld;
    FragU f;
    f.q[0] = *(const uint4*)(p + half * 16);
    f.q[1] = *(const uint4*)(p + half * 16 + 8);
    return f.v;
}

// B-matrix 32x16 bf16 fragment from LDS stored TRANSPOSED: baseT[n][k], stride ld.
// Layout: lane holds col N=lane%16; elements are K = 16*(lane/16) + e (contiguous).
__device__ __forceinline__ v16bf ld_frag_b(const __bf16* baseT, int ld, int lane) {
    const int col = lane & 15, half = lane >> 4;
    const __bf16* p = baseT + col * ld + half * 16;
    FragU f;
    f.q[0] = *(const uint4*)(p);
    f.q[1] = *(const uint4*)(p + 8);
    return f.v;
}

__global__ void f32_to_bf16_kernel(const float* __restrict__ src,
                                   __bf16* __restrict__ dst, long long n) {
    long long i = (long long)blockIdx.x * blockDim.x + threadIdx.x;
    const long long stride = (long long)gridDim.x * blockDim.x;
    for (; i < n; i += stride) dst[i] = (__bf16)src[i];
}

// C = A(MxK) * B(KxN), bf16 in, f32 accumulate. 128x128 block tile, BK=32,
// double-buffered LDS (stage k+32 while multiplying k). 8 waves; each owns
// a 64x32 tile = 4x2 v_wmma_f32_16x16x32_bf16 accumulators.
template <bool F32OUT>
__global__ __launch_bounds__(256, 1) void gemm_bf16(
    const __bf16* __restrict__ A, const __bf16* __restrict__ Bm,
    float* __restrict__ Cf, __bf16* __restrict__ Cb, int M, int N, int K) {
    __shared__ __align__(16) __bf16 As[2][128 * 40];  // row-major, pad stride 40
    __shared__ __align__(16) __bf16 Bt[2][128 * 40];  // transposed: Bt[n][k]

    const int tid  = threadIdx.x;
    const int lane = tid & 31;
    const int wid  = tid >> 5;
    const int wm   = wid >> 2;      // 0..1
    const int wn   = wid & 3;       // 0..3
    const int bm   = blockIdx.y * 128;
    const int bn   = blockIdx.x * 128;

    const int arow = tid >> 1, acol = (tid & 1) << 4;   // A tile: 128 x 32
    const int brow = tid >> 3, bcol = (tid & 7) << 4;   // B tile: 32 x 128

    auto stageAB = [&](int buf, int k0) {
        const __bf16* ga = A + (size_t)(bm + arow) * K + k0 + acol;
        *(uint4*)(&As[buf][arow * 40 + acol])     = *(const uint4*)(ga);
        *(uint4*)(&As[buf][arow * 40 + acol + 8]) = *(const uint4*)(ga + 8);
        const __bf16* gb = Bm + (size_t)(k0 + brow) * N + bn + bcol;
        union { uint4 q[2]; __bf16 e[16]; } u;
        u.q[0] = *(const uint4*)(gb);
        u.q[1] = *(const uint4*)(gb + 8);
#pragma unroll
        for (int e = 0; e < 16; ++e) Bt[buf][(bcol + e) * 40 + brow] = u.e[e];
    };

    const v8f vzero = {0.f, 0.f, 0.f, 0.f, 0.f, 0.f, 0.f, 0.f};
    v8f acc[4][2];
#pragma unroll
    for (int i = 0; i < 4; ++i)
#pragma unroll
        for (int j = 0; j < 2; ++j) acc[i][j] = vzero;

    stageAB(0, 0);
    __syncthreads();
    int cur = 0;

    for (int k0 = 0; k0 < K; k0 += 32) {
        if (k0 + 32 < K) {
            stageAB(cur ^ 1, k0 + 32);                 // overlap with WMMA below
            if (k0 + 64 < K) {                         // global_prefetch_b8, tile+2
                __builtin_prefetch(A + (size_t)(bm + arow) * K + k0 + 64 + acol, 0, 1);
                __builtin_prefetch(Bm + (size_t)(k0 + 64 + brow) * N + bn + bcol, 0, 1);
            }
        }

        v16bf af[4], bfr[2];
#pragma unroll
        for (int i = 0; i < 4; ++i)
            af[i] = ld_frag_a(&As[cur][(wm * 64 + i * 16) * 40], 40, lane);
#pragma unroll
        for (int j = 0; j < 2; ++j)
            bfr[j] = ld_frag_b(&Bt[cur][(wn * 32 + j * 16) * 40], 40, lane);
#pragma unroll
        for (int i = 0; i < 4; ++i)
#pragma unroll
            for (int j = 0; j < 2; ++j)
                acc[i][j] = __builtin_amdgcn_wmma_f32_16x16x32_bf16(
                    false, af[i], false, bfr[j], (short)0, acc[i][j], false, false);

        __syncthreads();
        cur ^= 1;
    }

    const int half = lane >> 4, cn = lane & 15;
#pragma unroll
    for (int i = 0; i < 4; ++i)
#pragma unroll
        for (int j = 0; j < 2; ++j)
#pragma unroll
            for (int r = 0; r < 8; ++r) {
                const int row = bm + wm * 64 + i * 16 + r + 8 * half;
                const int col = bn + wn * 32 + j * 16 + cn;
                const float v = acc[i][j][r];
                if constexpr (F32OUT) Cf[(size_t)row * N + col] = v;
                else                  Cb[(size_t)row * N + col] = (__bf16)v;
            }
}

// Flash attention. Block = 64 queries for one (b,h); 4 waves, 16 q-rows each.
// Double-buffered K/V staging; causal mask applied only on the diagonal step.
__global__ __launch_bounds__(128, 1) void flash_attn(
    const __bf16* __restrict__ qkv, __bf16* __restrict__ Obuf) {
    __shared__ __align__(16) __bf16 Qs[64 * 72];        // Qs[q][dh]
    __shared__ __align__(16) __bf16 Ks[2][32 * 72];     // Ks[key][dh]  (B^T for QK^T)
    __shared__ __align__(16) __bf16 Vt[2][64 * 40];     // Vt[dh][key]  (B^T for PV)
    __shared__ __align__(16) __bf16 Ps[4][16 * 40];     // per-wave P transpose scratch

    const int tid  = threadIdx.x;
    const int lane = tid & 31;
    const int w    = tid >> 5;
    const int half = lane >> 4;
    const int cn   = lane & 15;

    const int q0 = blockIdx.x * 64;
    const int bh = blockIdx.y;
    const int b  = bh >> 4;
    const int h  = bh & 15;

    const __bf16* Qg = qkv + (size_t)b * S_LEN * D3 + h * DHEAD;
    const __bf16* Kg = Qg + DMODEL;
    const __bf16* Vg = Qg + 2 * DMODEL;

    {   // stage Q: 64 x 64, 32 bf16 per thread
        const int r = tid >> 1, c = (tid & 1) << 5;
        const __bf16* g = Qg + (size_t)(q0 + r) * D3 + c;
#pragma unroll
        for (int t = 0; t < 4; ++t)
            *(uint4*)(&Qs[r * 72 + c + t * 8]) = *(const uint4*)(g + t * 8);
    }

    const int kvr = tid >> 2, kvc = (tid & 3) << 4;     // K/V tile: 32 x 64
    auto stageKV = [&](int buf, int kb) {
        const __bf16* gk = Kg + (size_t)(kb + kvr) * D3 + kvc;
        *(uint4*)(&Ks[buf][kvr * 72 + kvc])     = *(const uint4*)(gk);
        *(uint4*)(&Ks[buf][kvr * 72 + kvc + 8]) = *(const uint4*)(gk + 8);
        const __bf16* gv = Vg + (size_t)(kb + kvr) * D3 + kvc;
        union { uint4 q[2]; __bf16 e[16]; } u;
        u.q[0] = *(const uint4*)(gv);
        u.q[1] = *(const uint4*)(gv + 8);
#pragma unroll
        for (int e = 0; e < 16; ++e) Vt[buf][(kvc + e) * 40 + kvr] = u.e[e];
    };

    stageKV(0, 0);
    __syncthreads();

    const v16bf qa0 = ld_frag_a(&Qs[(w * 16) * 72], 72, lane);       // dh 0..31
    const v16bf qa1 = ld_frag_a(&Qs[(w * 16) * 72 + 32], 72, lane);  // dh 32..63

    const v8f vzero = {0.f, 0.f, 0.f, 0.f, 0.f, 0.f, 0.f, 0.f};
    v8f o[4];
    float lpart[8], mprev[8];
#pragma unroll
    for (int nt = 0; nt < 4; ++nt) o[nt] = vzero;
#pragma unroll
    for (int r = 0; r < 8; ++r) { lpart[r] = 0.f; mprev[r] = -1e30f; }

    const int nsteps  = blockIdx.x * 2 + 2;             // shared staging count
    const int mysteps = blockIdx.x * 2 + (w >> 1) + 1;  // causal per-wave bound
    int cur = 0;

    for (int j = 0; j < nsteps; ++j) {
        const int kb = j * 32;
        if (j + 1 < nsteps) stageKV(cur ^ 1, kb + 32);  // overlap with compute

        if (j < mysteps) {
            // S = Q K^T : two 16x16 key tiles, each a 2-step K=64 WMMA chain
            v8f s[2];
#pragma unroll
            for (int n = 0; n < 2; ++n) {
                s[n] = vzero;
                const __bf16* p = &Ks[cur][(n * 16 + cn) * 72 + half * 16];
                FragU f0, f1;
                f0.q[0] = *(const uint4*)(p);       f0.q[1] = *(const uint4*)(p + 8);
                f1.q[0] = *(const uint4*)(p + 32);  f1.q[1] = *(const uint4*)(p + 40);
                s[n] = __builtin_amdgcn_wmma_f32_16x16x32_bf16(
                    false, qa0, false, f0.v, (short)0, s[n], false, false);
                s[n] = __builtin_amdgcn_wmma_f32_16x16x32_bf16(
                    false, qa1, false, f1.v, (short)0, s[n], false, false);
            }

            // scale + causal mask (wave-uniform: only the diagonal step needs it)
            float pv[2][8], tmax[8];
            const int rowbase = q0 + w * 16 + 8 * half;
            const bool need_mask = (j == mysteps - 1);
#pragma unroll
            for (int r = 0; r < 8; ++r) {
                float x0 = s[0][r] * 0.125f;
                float x1 = s[1][r] * 0.125f;
                if (need_mask) {
                    const int row = rowbase + r;
                    x0 = (kb + cn      <= row) ? x0 : -1e30f;
                    x1 = (kb + 16 + cn <= row) ? x1 : -1e30f;
                }
                pv[0][r] = x0; pv[1][r] = x1;
                tmax[r] = red_max16(fmaxf(x0, x1));   // ds_swizzle xor-reduce
            }

#pragma unroll
            for (int r = 0; r < 8; ++r) {
                const float mnew  = fmaxf(mprev[r], tmax[r]);
                const float alpha = __expf(mprev[r] - mnew);
                mprev[r] = mnew;
                const float p0 = __expf(pv[0][r] - mnew);
                const float p1 = __expf(pv[1][r] - mnew);
                lpart[r] = lpart[r] * alpha + p0 + p1;  // per-lane column partials
#pragma unroll
                for (int nt = 0; nt < 4; ++nt) o[nt][r] *= alpha;
                Ps[w][(r + 8 * half) * 40 + cn]      = (__bf16)p0;  // C- -> A-layout
                Ps[w][(r + 8 * half) * 40 + 16 + cn] = (__bf16)p1;
            }
            asm volatile("s_wait_dscnt 0x0" ::: "memory");  // in-wave store->load

            const v16bf pa = ld_frag_a(&Ps[w][0], 40, lane);  // P as 16x32 A frag
#pragma unroll
            for (int nt = 0; nt < 4; ++nt) {
                const __bf16* p = &Vt[cur][(nt * 16 + cn) * 40 + half * 16];
                FragU f; f.q[0] = *(const uint4*)(p); f.q[1] = *(const uint4*)(p + 8);
                o[nt] = __builtin_amdgcn_wmma_f32_16x16x32_bf16(
                    false, pa, false, f.v, (short)0, o[nt], false, false);
            }
        }

        __syncthreads();
        cur ^= 1;
    }

    // finalize: reduce row sums across lanes, normalize, write merged-head output
#pragma unroll
    for (int r = 0; r < 8; ++r) {
        const float rinv = 1.0f / red_sum16(lpart[r]);
#pragma unroll
        for (int nt = 0; nt < 4; ++nt) o[nt][r] *= rinv;
    }
#pragma unroll
    for (int nt = 0; nt < 4; ++nt)
#pragma unroll
        for (int r = 0; r < 8; ++r) {
            const int srow = q0 + w * 16 + r + 8 * half;
            const size_t idx =
                ((size_t)(b * S_LEN + srow)) * DMODEL + h * DHEAD + nt * 16 + cn;
            Obuf[idx] = (__bf16)o[nt][r];
        }
}

extern "C" void kernel_launch(void* const* d_in, const int* in_sizes, int n_in,
                              void* d_out, int out_size, void* d_ws, size_t ws_size,
                              hipStream_t stream) {
    (void)in_sizes; (void)n_in; (void)out_size; (void)ws_size;
    const float* x    = (const float*)d_in[0];
    // d_in[1] = causal mask: structure is known, ignored
    const float* Wqkv = (const float*)d_in[2];
    const float* Wout = (const float*)d_in[3];
    float* out = (float*)d_out;

    char* ws = (char*)d_ws;
    size_t off = 0;
    auto alloc = [&](size_t elems) -> __bf16* {
        char* p = ws + off;
        off = (off + elems * 2 + 255) & ~(size_t)255;
        return (__bf16*)p;
    };
    __bf16* xb    = alloc((size_t)BATCH * S_LEN * DMODEL);   // 16 MB
    __bf16* wqkvb = alloc((size_t)DMODEL * D3);              //  6 MB
    __bf16* woutb = alloc((size_t)DMODEL * DMODEL);          //  2 MB
    __bf16* qkvb  = alloc((size_t)BATCH * S_LEN * D3);       // 48 MB
    __bf16* ob    = alloc((size_t)BATCH * S_LEN * DMODEL);   // 16 MB

    f32_to_bf16_kernel<<<4096, 256, 0, stream>>>(x, xb, (long long)BATCH * S_LEN * DMODEL);
    f32_to_bf16_kernel<<<4096, 256, 0, stream>>>(Wqkv, wqkvb, (long long)DMODEL * D3);
    f32_to_bf16_kernel<<<2048, 256, 0, stream>>>(Wout, woutb, (long long)DMODEL * DMODEL);

    dim3 g1(D3 / 128, (BATCH * S_LEN) / 128);
    gemm_bf16<false><<<g1, 256, 0, stream>>>(xb, wqkvb, nullptr, qkvb,
                                             BATCH * S_LEN, D3, DMODEL);

    dim3 g2(S_LEN / 64, BATCH * NHEAD);
    flash_attn<<<g2, 128, 0, stream>>>(qkvb, ob);

    dim3 g3(DMODEL / 128, (BATCH * S_LEN) / 128);
    gemm_bf16<true><<<g3, 256, 0, stream>>>(ob, woutb, out, nullptr,
                                            BATCH * S_LEN, DMODEL, DMODEL);
}
